// DotProductAttention_11433202942822
// MI455X (gfx1250) — matrix operand order repaired and checked by reading, not compiled, not run on hardware
//
#include <hip/hip_runtime.h>
#include <stdint.h>

// MI455X / gfx1250 flash-attention forward.
// bf16 WMMA (f32 accum) + TDM tensor_load_to_lds for the mask tile,
// double-buffered one tile ahead, completed with s_wait_tensorcnt.

typedef __attribute__((ext_vector_type(16))) __bf16    v16bf;
typedef __attribute__((ext_vector_type(8)))  float     v8f;
typedef __attribute__((ext_vector_type(4)))  uint32_t  v4u;
typedef __attribute__((ext_vector_type(8)))  int       v8i;
typedef __attribute__((ext_vector_type(4)))  int       v4i;

#define NBATCH 2
#define NHEAD  16
#define TSEQ   2048
#define DHEAD  64
#define KT     64               // keys per iteration
#define QW     16               // queries per wave
#define NWAVES 8
#define QB     (QW * NWAVES)    // 128 queries per block
#define NITER  (TSEQ / KT)      // 32
#define SCALE  0.125f           // 1/sqrt(64)

#if __has_builtin(__builtin_amdgcn_tensor_load_to_lds) && \
    __has_builtin(__builtin_amdgcn_s_wait_tensorcnt)
#define HAVE_TDM 1
#else
#define HAVE_TDM 0
#endif

union Frag {
    v16bf v;
    uint4 q[2];
    unsigned short u[16];
};

static __device__ __forceinline__ unsigned short f2bf(float f) {
    union { __bf16 b; unsigned short u; } p;
    p.b = (__bf16)f;            // native f32->bf16 cvt
    return p.u;
}
static __device__ __forceinline__ uint32_t pk2bf(float lo, float hi) {
    return (uint32_t)f2bf(lo) | ((uint32_t)f2bf(hi) << 16);
}

#if HAVE_TDM
// TDM: load a [rows=QB] x [cols=KT] f32 tile of the (TSEQ x TSEQ) mask into LDS.
// D# per CDNA5 ISA ch.8: group0 = {flags, lds_addr, gaddr_lo, gaddr_hi|type},
// group1 packs data_size / dims / tile dims / dim0 stride.
// This toolchain exposes the 6-arg builtin: (g0, g1, g2, g3, g4, cpol).
static __device__ __forceinline__ void tdm_load_mask_tile(const float* gaddr,
                                                          uint32_t lds_off) {
    uint64_t ga = (uint64_t)(uintptr_t)gaddr;
    v4u g0;
    g0[0] = 1u;                                   // count=1, user descriptor
    g0[1] = lds_off;                              // LDS byte address
    g0[2] = (uint32_t)ga;                         // global addr lo
    g0[3] = (uint32_t)(ga >> 32) | (2u << 30);    // global addr hi | type=2
    v8i g1;
    g1[0] = (int)(2u << 16);                      // data_size = 2 (4 bytes)
    g1[1] = (int)((TSEQ & 0xFFFFu) << 16);        // tensor_dim0 lo16
    g1[2] = (int)(((TSEQ >> 16) & 0xFFFFu) | ((TSEQ & 0xFFFFu) << 16)); // d0 hi | d1 lo
    g1[3] = (int)(((TSEQ >> 16) & 0xFFFFu) | ((uint32_t)KT << 16));     // d1 hi | tile_dim0
    g1[4] = (int)QB;                              // tile_dim1=128, tile_dim2=0
    g1[5] = (int)TSEQ;                            // tensor_dim0_stride lo32
    g1[6] = 0;                                    // stride hi | dim1_stride lo
    g1[7] = 0;
    v4i g2 = {0, 0, 0, 0};
    v4i g3 = {0, 0, 0, 0};
    v8i g4 = {0, 0, 0, 0, 0, 0, 0, 0};
    __builtin_amdgcn_tensor_load_to_lds(g0, g1, g2, g3, g4, 0);
}
#endif

__global__ __launch_bounds__(256, 1) void
fa_fwd_kernel(const float* __restrict__ Qg, const float* __restrict__ Kg,
              const float* __restrict__ Vg, const float* __restrict__ Mg,
              const int* __restrict__ isCp, float* __restrict__ Og)
{
    __shared__ __align__(16) unsigned short klds[KT * DHEAD];       // K tile [key][d]   bf16, 8KB
    __shared__ __align__(16) unsigned short vlds[DHEAD * KT];       // V tile [d][key]   bf16, 8KB
    __shared__ __align__(16) unsigned short plds[NWAVES][QW * KT];  // P staging, 16KB
    __shared__ __align__(16) float          mlds[2][QB * KT];       // mask tiles, 64KB

    const int tid  = threadIdx.x;
    const int wv   = tid >> 5;
    const int lane = tid & 31;
    const int hw   = lane >> 4;
    const int ln   = lane & 15;

    const int bh = blockIdx.x >> 4;      // b*16+h
    const int qt = blockIdx.x & 15;
    const int b  = bh >> 4;

    const size_t headOff = (size_t)bh * TSEQ * DHEAD;
    const float* Qb = Qg + headOff;
    const float* Kb = Kg + headOff;
    const float* Vb = Vg + headOff;
    float*       Ob = Og + headOff;
    const float* Mb = Mg + (size_t)b * TSEQ * TSEQ + (size_t)(qt * QB) * TSEQ;

    const int causal = isCp[0];
    const int qbase  = qt * QB + wv * QW;

    // ---- Q fragments (A-matrix 16-bit 16x32 layout), pre-scaled ----
    Frag aq[2];
    {
        const float* qp = Qb + (size_t)(qbase + ln) * DHEAD;
#pragma unroll
        for (int g = 0; g < 2; ++g)
#pragma unroll
            for (int e = 0; e < 16; ++e) {
                int d = g * 32 + hw * 16 + e;
                aq[g].u[e] = f2bf(qp[d] * SCALE);
            }
    }

    v8f o[4];
#pragma unroll
    for (int t = 0; t < 4; ++t) o[t] = (v8f){0.f,0.f,0.f,0.f,0.f,0.f,0.f,0.f};
    float mrow[8], lrow[8];
#pragma unroll
    for (int r = 0; r < 8; ++r) { mrow[r] = -1e30f; lrow[r] = 0.0f; }

#if HAVE_TDM
    if (wv == 0)   // one wave drives the DMA; TDM ignores EXEC
        tdm_load_mask_tile(Mb, (uint32_t)(uintptr_t)&mlds[0][0]);
#endif

    for (int it = 0; it < NITER; ++it) {
        const int kb  = it * KT;
        const int cur = it & 1;

#if HAVE_TDM
        if (wv == 0 && (it + 1) < NITER)    // prefetch next mask tile (double buffer)
            tdm_load_mask_tile(Mb + (size_t)(kb + KT),
                               (uint32_t)(uintptr_t)&mlds[(it + 1) & 1][0]);
#endif

        // ---- cooperative stage: K (row-major) / V (transposed) -> LDS bf16 ----
        {
            int key = tid >> 2;              // 0..63
            int d0  = (tid & 3) * 16;        // 0/16/32/48
            const float* kp = Kb + (size_t)(kb + key) * DHEAD + d0;
            float4 k0 = ((const float4*)kp)[0];
            float4 k1 = ((const float4*)kp)[1];
            float4 k2 = ((const float4*)kp)[2];
            float4 k3 = ((const float4*)kp)[3];
            uint4 pa, pb;
            pa.x = pk2bf(k0.x, k0.y); pa.y = pk2bf(k0.z, k0.w);
            pa.z = pk2bf(k1.x, k1.y); pa.w = pk2bf(k1.z, k1.w);
            pb.x = pk2bf(k2.x, k2.y); pb.y = pk2bf(k2.z, k2.w);
            pb.z = pk2bf(k3.x, k3.y); pb.w = pk2bf(k3.z, k3.w);
            *(uint4*)&klds[key * DHEAD + d0]     = pa;
            *(uint4*)&klds[key * DHEAD + d0 + 8] = pb;

            const float* vp = Vb + (size_t)(kb + key) * DHEAD + d0;
            float4 w0 = ((const float4*)vp)[0];
            float4 w1 = ((const float4*)vp)[1];
            float4 w2 = ((const float4*)vp)[2];
            float4 w3 = ((const float4*)vp)[3];
            float vv[16] = { w0.x,w0.y,w0.z,w0.w, w1.x,w1.y,w1.z,w1.w,
                             w2.x,w2.y,w2.z,w2.w, w3.x,w3.y,w3.z,w3.w };
#pragma unroll
            for (int i = 0; i < 16; ++i)
                vlds[(d0 + i) * KT + key] = f2bf(vv[i]);

            if (kb + KT < TSEQ) {            // prefetch next K/V tile
                __builtin_prefetch(kp + (size_t)KT * DHEAD, 0, 1);
                __builtin_prefetch(vp + (size_t)KT * DHEAD, 0, 1);
            }
        }

#if HAVE_TDM
        if (wv == 0) {                       // current mask tile must have landed
            if ((it + 1) < NITER) __builtin_amdgcn_s_wait_tensorcnt((short)1);
            else                  __builtin_amdgcn_s_wait_tensorcnt((short)0);
        }
#else
        {   // fallback: cooperative mask load (4 x b128 per thread)
            int row = tid >> 1;              // 0..127
            int c0  = (tid & 1) * 32;
            const float* mp = Mb + (size_t)row * TSEQ + kb + c0;
            float4* dst = (float4*)&mlds[cur][row * KT + c0];
            dst[0] = ((const float4*)mp)[0];
            dst[1] = ((const float4*)mp)[1];
            dst[2] = ((const float4*)mp)[2];
            dst[3] = ((const float4*)mp)[3];
        }
#endif
        __syncthreads();

        // ---- scores: S[16q x 64k] as four 16x16 C tiles, 2 WMMAs each over D ----
        v8f s[4];
#pragma unroll
        for (int j = 0; j < 4; ++j) {
            v8f c = (v8f){0.f,0.f,0.f,0.f,0.f,0.f,0.f,0.f};
#pragma unroll
            for (int g = 0; g < 2; ++g) {
                Frag bk;
                const uint4* kp = (const uint4*)&klds[(j * 16 + ln) * DHEAD + g * 32 + hw * 16];
                bk.q[0] = kp[0]; bk.q[1] = kp[1];
                c = __builtin_amdgcn_wmma_f32_16x16x32_bf16(
                        false, aq[g].v, false, bk.v, (short)0, c, false, false);
            }
            s[j] = c;
        }

        // ---- mask / causal from LDS tile ----
#pragma unroll
        for (int j = 0; j < 4; ++j) {
            int kidx = kb + j * 16 + ln;
#pragma unroll
            for (int r = 0; r < 8; ++r) {
                int m = r + 8 * hw;
                float mv = mlds[cur][(wv * QW + m) * KT + j * 16 + ln];
                int q = qbase + m;
                if (mv == 0.0f || (causal && kidx > q)) s[j][r] = -__builtin_inff();
            }
        }

        // ---- online softmax; P written straight into A-fragment-ordered LDS ----
        float alpha[8];
#pragma unroll
        for (int r = 0; r < 8; ++r) {
            float rm = fmaxf(fmaxf(s[0][r], s[1][r]), fmaxf(s[2][r], s[3][r]));
            rm = fmaxf(rm, __shfl_xor(rm, 1));
            rm = fmaxf(rm, __shfl_xor(rm, 2));
            rm = fmaxf(rm, __shfl_xor(rm, 4));
            rm = fmaxf(rm, __shfl_xor(rm, 8));
            float mn = fmaxf(mrow[r], rm);
            alpha[r] = __expf(mrow[r] - mn);
            mrow[r]  = mn;
            float rs = 0.0f;
            int m = r + 8 * hw;
#pragma unroll
            for (int j = 0; j < 4; ++j) {
                float pv = __expf(s[j][r] - mn);
                rs += pv;
                int k  = j * 16 + ln;         // key within 64-tile
                int f  = k >> 5;              // which 16x32 A fragment
                int kl = k & 31;
                int Lf = m + 16 * (kl >> 4);
                int e  = kl & 15;
                plds[wv][f * 512 + Lf * 16 + e] = f2bf(pv);
            }
            rs += __shfl_xor(rs, 1);
            rs += __shfl_xor(rs, 2);
            rs += __shfl_xor(rs, 4);
            rs += __shfl_xor(rs, 8);
            lrow[r] = lrow[r] * alpha[r] + rs;
        }
#pragma unroll
        for (int t = 0; t < 4; ++t)
#pragma unroll
            for (int r = 0; r < 8; ++r) o[t][r] *= alpha[r];

        asm volatile("s_wait_dscnt 0" ::: "memory");   // in-wave LDS RAW fence

        Frag pf[2];
#pragma unroll
        for (int f = 0; f < 2; ++f) {
            const uint4* pp = (const uint4*)&plds[wv][f * 512 + lane * 16];
            pf[f].q[0] = pp[0]; pf[f].q[1] = pp[1];
        }

        // ---- O += P x V : 64-key contraction = 2 A-frags x 4 d-tiles ----
#pragma unroll
        for (int t = 0; t < 4; ++t) {
            v8f acc = o[t];
#pragma unroll
            for (int f = 0; f < 2; ++f) {
                Frag bv;
                const uint4* vp = (const uint4*)&vlds[(t * 16 + ln) * KT + f * 32 + hw * 16];
                bv.q[0] = vp[0]; bv.q[1] = vp[1];
                acc = __builtin_amdgcn_wmma_f32_16x16x32_bf16(
                          false, pf[f].v, false, bv.v, (short)0, acc, false, false);
            }
            o[t] = acc;
        }
        __syncthreads();
    }

    // ---- epilogue: normalize and store ----
    float inv[8];
#pragma unroll
    for (int r = 0; r < 8; ++r) inv[r] = 1.0f / lrow[r];
#pragma unroll
    for (int t = 0; t < 4; ++t)
#pragma unroll
        for (int r = 0; r < 8; ++r) {
            int q = qbase + r + 8 * hw;
            int d = t * 16 + ln;
            Ob[(size_t)q * DHEAD + d] = o[t][r] * inv[r];
        }
}

extern "C" void kernel_launch(void* const* d_in, const int* in_sizes, int n_in,
                              void* d_out, int out_size, void* d_ws, size_t ws_size,
                              hipStream_t stream) {
    (void)in_sizes; (void)n_in; (void)out_size; (void)d_ws; (void)ws_size;
    const float* Q = (const float*)d_in[0];
    const float* K = (const float*)d_in[1];
    const float* V = (const float*)d_in[2];
    const float* M = (const float*)d_in[3];
    const int*   C = (const int*)  d_in[4];
    float* O = (float*)d_out;
    dim3 grid(NBATCH * NHEAD * (TSEQ / QB));   // 512 blocks
    dim3 block(256);                            // 8 wave32s
    fa_fwd_kernel<<<grid, block, 0, stream>>>(Q, K, V, M, C, O);
}
